// TimeSeriesMambaRegressor_86706799772354
// MI455X (gfx1250) — matrix operand order repaired and checked by reading, not compiled, not run on hardware
//
#include <hip/hip_runtime.h>
#include <cmath>

// ---- problem constants (from reference) ----
#define BATCH 8
#define SEQL  2048
#define LP    2049          // L + 1 (prepended embedding token)
#define FDIM  32
#define DMOD  256
#define NLAY  2
#define DIN   512           // DI = 2*DM
#define DST   16            // DS
#define DCV   4
#define RKK   16            // RK = DM/16
#define XPN   (RKK + 2*DST) // 48

typedef __attribute__((ext_vector_type(16))) _Float16 v16h;
typedef __attribute__((ext_vector_type(8)))  float    v8f;

__device__ __forceinline__ float4 ld4(const float* p) {
    return *reinterpret_cast<const float4*>(p);
}
__device__ __forceinline__ void cv4(v16h& dst, int base, float4 v) {
    dst[base + 0] = (_Float16)v.x; dst[base + 1] = (_Float16)v.y;
    dst[base + 2] = (_Float16)v.z; dst[base + 3] = (_Float16)v.w;
}

// -----------------------------------------------------------------------------
// WMMA GEMM:  C[M,N] = act( A[M,K] @ W[N,K]^T + bias[N] )
// A row-major (stride LDA), W row-major (N x K), C row-major (stride LDC).
// One wave computes TN adjacent 16x16 output tiles (A fragment reused TN times).
// K/LDA/LDC compile-time -> all addresses affine with immediate offsets.
// M tail handled by clamped loads + one uniform branch in the epilogue (WMMA
// row m of D depends only on row m of A, so clamped garbage rows are simply
// never stored).  Grid must cover N exactly (all our N are multiples of 16*TN).
// ACT: 0 none, 1 softplus.
// REMAP: 1 = feature-proj row remap m -> m + (m0>>11) + 1  (tile never crosses
//        a batch boundary because 2048 % 16 == 0).
// -----------------------------------------------------------------------------
template <int K, int LDA, int LDC, int TN, int ACT, int REMAP>
__global__ __launch_bounds__(32)
void wmma_gemm(const float* __restrict__ A,
               const float* __restrict__ W,
               const float* __restrict__ bias,
               float* __restrict__ C,
               int M, int N)
{
    const int l     = threadIdx.x;       // lane 0..31
    const int lhalf = l >> 4;            // 0 or 1
    const int mr    = l & 15;
    const int nc    = l & 15;
    const int m0    = blockIdx.y * 16;
    const int n0    = blockIdx.x * (16 * TN);

    // clamped A row -> unconditional vector loads, no exec manipulation
    const int arow = min(m0 + mr, M - 1);
    const float* Ap = A + (size_t)arow * LDA;

    const float* Wp[TN];
#pragma unroll
    for (int tn = 0; tn < TN; ++tn) {
        int bcol = min(n0 + tn * 16 + nc, N - 1);
        Wp[tn] = W + (size_t)bcol * K;
    }

    v8f acc[TN];
#pragma unroll
    for (int tn = 0; tn < TN; ++tn) acc[tn] = {};

    constexpr int KSTEPS = (K + 31) / 32;
#pragma unroll
    for (int ks = 0; ks < KSTEPS; ++ks) {
        constexpr int dummy = 0; (void)dummy;
        const int k0 = ks * 32;
        v16h a;
        // A fragment: lane<16 holds K {k0..k0+7, k0+16..k0+23}; lane>=16 shifts by 8
        cv4(a, 0, ld4(Ap + k0 + lhalf * 8));
        cv4(a, 4, ld4(Ap + k0 + lhalf * 8 + 4));
        if (K % 32 == 0 || k0 + 16 < K) {            // folds at compile time
            cv4(a,  8, ld4(Ap + k0 + 16 + lhalf * 8));
            cv4(a, 12, ld4(Ap + k0 + 16 + lhalf * 8 + 4));
        } else {
#pragma unroll
            for (int e = 8; e < 16; ++e) a[e] = (_Float16)0.0f;
        }
#pragma unroll
        for (int tn = 0; tn < TN; ++tn) {
            v16h b;
            // B fragment: lane = column; lane<16 holds K {k0..k0+7, k0+16..k0+23}; lane>=16 shifts by 8
            const float* wp = Wp[tn] + k0 + lhalf * 8;
            cv4(b, 0, ld4(wp + 0));
            cv4(b, 4, ld4(wp + 4));
            if (K % 32 == 0 || k0 + 16 < K) {
                cv4(b,  8, ld4(wp + 16));
                cv4(b, 12, ld4(wp + 20));
            } else {
#pragma unroll
                for (int e = 8; e < 16; ++e) b[e] = (_Float16)0.0f;
            }
            acc[tn] = __builtin_amdgcn_wmma_f32_16x16x32_f16(
                false, a, false, b, (short)0, acc[tn], false, false);
        }
    }

    // epilogue: D layout -> lane half selects rows 0..7 / 8..15, lane%16 = col.
    const int  row0  = m0 + lhalf * 8;
    const int  orow0 = REMAP ? row0 + (m0 >> 11) + 1 : row0;   // affine remap
    const bool fullM = (m0 + 16 <= M);
#pragma unroll
    for (int tn = 0; tn < TN; ++tn) {
        const int   bcol = n0 + tn * 16 + nc;
        const float bv   = bias ? bias[bcol] : 0.0f;
        float* Cp = C + (size_t)orow0 * LDC + bcol;
        if (fullM) {
#pragma unroll
            for (int r = 0; r < 8; ++r) {
                float v = acc[tn][r] + bv;
                if (ACT == 1) v = (v > 20.0f) ? v : log1pf(__expf(v));  // softplus
                Cp[r * LDC] = v;                 // immediate offsets
            }
        } else {
#pragma unroll
            for (int r = 0; r < 8; ++r) {
                if (row0 + r < M) {
                    float v = acc[tn][r] + bv;
                    if (ACT == 1) v = (v > 20.0f) ? v : log1pf(__expf(v));
                    Cp[r * LDC] = v;
                }
            }
        }
    }
}

// ---- write embedding token into row 0 of every batch ----
__global__ void embed_kernel(const float* __restrict__ emb,
                             const int* __restrict__ loc_ids,
                             float* __restrict__ x)
{
    int b = blockIdx.x, d = threadIdx.x;                 // grid 8, block 256
    x[((size_t)b * LP) * DMOD + d] = emb[(size_t)loc_ids[b] * DMOD + d];
}

// ---- depthwise causal conv (DC=4) + bias + SiLU on xin = xz[..., :DI] ----
__global__ void conv_silu_kernel(const float* __restrict__ xz,
                                 const float* __restrict__ cw,   // (DI, DC)
                                 const float* __restrict__ cb,   // (DI,)
                                 float* __restrict__ xc)
{
    size_t i = (size_t)blockIdx.x * blockDim.x + threadIdx.x;
    if (i >= (size_t)BATCH * LP * DIN) return;
    int d   = (int)(i % DIN);
    size_t row = i / DIN;
    int t = (int)(row % LP);
    size_t brow = row - t;                               // b*LP
    float acc = cb[d];
#pragma unroll
    for (int k = 0; k < DCV; ++k) {
        int ts = t + k - (DCV - 1);
        if (ts >= 0)
            acc += xz[(brow + ts) * 1024 + d] * cw[d * DCV + k];
    }
    xc[i] = acc / (1.0f + __expf(-acc));                 // SiLU
}

// ---- selective scan: one thread per (b, d); 16-wide state in registers ----
__global__ __launch_bounds__(256)
void scan_kernel(const float* __restrict__ dt,     // (B, LP, DI)
                 const float* __restrict__ xdbl,   // (B, LP, 48)  [dt|B|C]
                 const float* __restrict__ xc,     // (B, LP, DI)
                 const float* __restrict__ A_log,  // (DI, DS)
                 float* __restrict__ ys)           // (B, LP, DI)
{
    int d = blockIdx.x * blockDim.x + threadIdx.x;       // grid.x = DI/256
    int b = blockIdx.y;
    float Ad[DST];
#pragma unroll
    for (int s = 0; s < DST; ++s) Ad[s] = -__expf(A_log[d * DST + s]);
    float h[DST];
#pragma unroll
    for (int s = 0; s < DST; ++s) h[s] = 0.0f;

    size_t base = (size_t)b * LP;
    for (int t = 0; t < LP; ++t) {
        size_t r = base + t;
        float dtv = dt[r * DIN + d];
        float u   = xc[r * DIN + d];
        float du  = dtv * u;
        const float* xr = xdbl + r * XPN;
        float4 Bq[4], Cq[4];
#pragma unroll
        for (int q = 0; q < 4; ++q) {
            Bq[q] = ld4(xr + RKK + 4 * q);
            Cq[q] = ld4(xr + RKK + DST + 4 * q);
        }
        const float* Bv = reinterpret_cast<const float*>(Bq);
        const float* Cv = reinterpret_cast<const float*>(Cq);
        float y = 0.0f;
#pragma unroll
        for (int s = 0; s < DST; ++s) {
            h[s] = __expf(dtv * Ad[s]) * h[s] + du * Bv[s];
            y += h[s] * Cv[s];
        }
        ys[r * DIN + d] = y;
    }
}

// ---- y = (ys + D*xc) * silu(z);  z = xz[..., DI + d];  writes into xc ----
__global__ void gate_kernel(const float* __restrict__ ys,
                            const float* __restrict__ xz,
                            const float* __restrict__ Dp,
                            float* __restrict__ xc)
{
    size_t i = (size_t)blockIdx.x * blockDim.x + threadIdx.x;
    if (i >= (size_t)BATCH * LP * DIN) return;
    int d = (int)(i % DIN);
    size_t row = i / DIN;
    float z = xz[row * 1024 + DIN + d];
    float sz = z / (1.0f + __expf(-z));
    xc[i] = (ys[i] + Dp[d] * xc[i]) * sz;
}

// ---- layernorm(last token) -> GELU MLP -> scalar; one block per batch ----
__global__ __launch_bounds__(256)
void head_kernel(const float* __restrict__ x,
                 const float* __restrict__ g, const float* __restrict__ bta,
                 const float* __restrict__ h1w, const float* __restrict__ h1b,
                 const float* __restrict__ h2w, const float* __restrict__ h2b,
                 float* __restrict__ out)
{
    __shared__ float sh[DMOD];
    __shared__ float red[DMOD];
    int b = blockIdx.x, tid = threadIdx.x;
    const float* row = x + ((size_t)b * LP + (LP - 1)) * DMOD;
    float v = row[tid];

    red[tid] = v; __syncthreads();
    for (int s = 128; s > 0; s >>= 1) { if (tid < s) red[tid] += red[tid + s]; __syncthreads(); }
    float mean = red[0] / (float)DMOD; __syncthreads();

    float dv = v - mean;
    red[tid] = dv * dv; __syncthreads();
    for (int s = 128; s > 0; s >>= 1) { if (tid < s) red[tid] += red[tid + s]; __syncthreads(); }
    float var = red[0] / (float)DMOD; __syncthreads();

    sh[tid] = dv * rsqrtf(var + 1e-5f) * g[tid] + bta[tid];
    __syncthreads();

    float acc = h1b[tid];
    for (int i = 0; i < DMOD; ++i) acc += sh[i] * h1w[tid * DMOD + i];
    float ge = 0.5f * acc * (1.0f + erff(acc * 0.70710678f));   // exact GELU

    red[tid] = ge * h2w[tid]; __syncthreads();
    for (int s = 128; s > 0; s >>= 1) { if (tid < s) red[tid] += red[tid + s]; __syncthreads(); }
    if (tid == 0) out[b] = red[0] + h2b[0];
}

// -----------------------------------------------------------------------------
extern "C" void kernel_launch(void* const* d_in, const int* in_sizes, int n_in,
                              void* d_out, int out_size, void* d_ws, size_t ws_size,
                              hipStream_t stream)
{
    const float* x_seq     = (const float*)d_in[0];
    const int*   loc_ids   = (const int*)  d_in[1];
    const float* feat_w    = (const float*)d_in[2];
    const float* feat_b    = (const float*)d_in[3];
    const float* emb       = (const float*)d_in[4];
    const float* in_proj_w = (const float*)d_in[5];
    const float* conv_w    = (const float*)d_in[6];
    const float* conv_b    = (const float*)d_in[7];
    const float* x_proj_w  = (const float*)d_in[8];
    const float* dt_proj_w = (const float*)d_in[9];
    const float* dt_proj_b = (const float*)d_in[10];
    const float* A_log     = (const float*)d_in[11];
    const float* Dp        = (const float*)d_in[12];
    const float* out_proj_w= (const float*)d_in[13];
    const float* norm_g    = (const float*)d_in[14];
    const float* norm_b    = (const float*)d_in[15];
    const float* h1_w      = (const float*)d_in[16];
    const float* h1_b      = (const float*)d_in[17];
    const float* h2_w      = (const float*)d_in[18];
    const float* h2_b      = (const float*)d_in[19];

    const int M  = BATCH * LP;            // 16392 rows
    const int MT = (M + 15) / 16;         // 1025 row tiles

    // workspace carve-up (floats)
    float* ws   = (float*)d_ws;
    size_t off  = 0;
    float* x    = ws + off; off += (size_t)M * DMOD;        // (B,LP,DM)
    float* xz   = ws + off; off += (size_t)M * 2 * DIN;     // (B,LP,1024)
    float* xc   = ws + off; off += (size_t)M * DIN;         // (B,LP,DI) (also y)
    float* xdbl = ws + off; off += (size_t)M * XPN;         // (B,LP,48)
    float* dtb  = ws + off; off += (size_t)M * DIN;         // (B,LP,DI)
    float* ysb  = ws + off; off += (size_t)M * DIN;         // (B,LP,DI)
    (void)ws_size; (void)in_sizes; (void)n_in; (void)out_size;

    const size_t nEl = (size_t)M * DIN;
    const int    ew  = (int)((nEl + 255) / 256);

    // 1) embedding token + feature projection (remapped rows)
    embed_kernel<<<dim3(BATCH), dim3(DMOD), 0, stream>>>(emb, loc_ids, x);
    wmma_gemm<FDIM, FDIM, DMOD, 4, 0, 1>
        <<<dim3(DMOD / 64, (BATCH * SEQL) / 16), dim3(32), 0, stream>>>(
        x_seq, feat_w, feat_b, x, BATCH * SEQL, DMOD);

    for (int l = 0; l < NLAY; ++l) {
        const float* ipw = in_proj_w + (size_t)l * 2 * DIN * DMOD;
        const float* cw  = conv_w    + (size_t)l * DIN * DCV;
        const float* cb  = conv_b    + (size_t)l * DIN;
        const float* xpw = x_proj_w  + (size_t)l * XPN * DIN;
        const float* dpw = dt_proj_w + (size_t)l * DIN * RKK;
        const float* dpb = dt_proj_b + (size_t)l * DIN;
        const float* Al  = A_log     + (size_t)l * DIN * DST;
        const float* Dl  = Dp        + (size_t)l * DIN;
        const float* opw = out_proj_w+ (size_t)l * DMOD * DIN;

        // xz = x @ in_proj_w^T                (16392 x 1024 x 256)
        wmma_gemm<DMOD, DMOD, 2 * DIN, 4, 0, 0>
            <<<dim3((2 * DIN) / 64, MT), dim3(32), 0, stream>>>(
            x, ipw, nullptr, xz, M, 2 * DIN);

        // xc = silu(conv(xin) + cb)
        conv_silu_kernel<<<dim3(ew), dim3(256), 0, stream>>>(xz, cw, cb, xc);

        // xdbl = xc @ x_proj_w^T              (16392 x 48 x 512), N=48 -> TN=3
        wmma_gemm<DIN, DIN, XPN, 3, 0, 0>
            <<<dim3(1, MT), dim3(32), 0, stream>>>(
            xc, xpw, nullptr, xdbl, M, XPN);

        // dt = softplus(xdbl[...,:RK] @ dt_proj_w^T + dpb)   (K=16 zero-padded)
        wmma_gemm<RKK, XPN, DIN, 4, 1, 0>
            <<<dim3(DIN / 64, MT), dim3(32), 0, stream>>>(
            xdbl, dpw, dpb, dtb, M, DIN);

        // sequential selective scan
        scan_kernel<<<dim3(DIN / 256, BATCH), dim3(256), 0, stream>>>(
            dtb, xdbl, xc, Al, ysb);

        // y = (ys + D*xc) * silu(z)   (in place into xc)
        gate_kernel<<<dim3(ew), dim3(256), 0, stream>>>(ysb, xz, Dl, xc);

        // x = y @ out_proj_w^T                (16392 x 256 x 512)
        wmma_gemm<DIN, DIN, DMOD, 4, 0, 0>
            <<<dim3(DMOD / 64, MT), dim3(32), 0, stream>>>(
            xc, opw, nullptr, x, M, DMOD);
    }

    // layernorm + GELU MLP head
    head_kernel<<<dim3(BATCH), dim3(DMOD), 0, stream>>>(
        x, norm_g, norm_b, h1_w, h1_b, h2_w, h2_b, (float*)d_out);
}
